// InputFeedRNNDecoderLayers_68204080661055
// MI455X (gfx1250) — matrix operand order repaired and checked by reading, not compiled, not run on hardware
//
#include <hip/hip_runtime.h>
#include <hip/hip_bf16.h>
#include <math.h>
#include <stdint.h>

// ---------------------------------------------------------------------------
// Problem constants (from reference): T=64, B=64, S=64, E=1024, D=1024, L=2
// ---------------------------------------------------------------------------
#define TT 64
#define BB 64
#define SS 64
#define DD 1024

typedef __attribute__((ext_vector_type(16))) __bf16 v16bf;
typedef __attribute__((ext_vector_type(8)))  float  v8f;
typedef __attribute__((ext_vector_type(4)))  unsigned int u32x4;

__device__ __forceinline__ void wait_async0() {
#if __has_builtin(__builtin_amdgcn_s_wait_asynccnt)
    __builtin_amdgcn_s_wait_asynccnt(0);
#else
    asm volatile("s_wait_asynccnt 0x0" ::: "memory");
#endif
}

// ---------------------------------------------------------------------------
// WMMA GEMM:  Y[64, N] = X[64, K](bf16, row-major, ldx) * Wt[K, N](bf16, ldw)
// Block = 128 threads (4 waves). Block covers all M=64 rows of a 64-wide
// N-strip; wave w owns m-tile rows 16w..16w+15. The 32x64 B slab for each
// K-step is staged once into LDS with async global->LDS DMA (double
// buffered, ASYNCcnt), shared by all 4 waves, then consumed via strided 16-bit LDS reads
// into the ISA B-fragment layout. grid = (N/64), block = 128.
// ---------------------------------------------------------------------------
#define BROW 72   // 64 N elements + 8 pad (keeps 16B alignment, breaks 2^k stride)

__global__ __launch_bounds__(128)
void wmma_gemm_kernel(const __hip_bfloat16* __restrict__ X, int ldx,
                      const __hip_bfloat16* __restrict__ Wt, int ldw,
                      float* __restrict__ Y, int ldy, int K) {
    __shared__ __hip_bfloat16 Bsm[2][32 * BROW];

    const int tid  = threadIdx.x;          // 0..127
    const int lane = tid & 31;             // wave32 lane
    const int wave = tid >> 5;             // m-tile index
    const int n0   = blockIdx.x * 64;
    const int m0   = wave * 16;

    // staging map: 32 rows x 128B; each thread DMAs 32 contiguous bytes
    const int srow = tid >> 2;             // 0..31 (K-row within slab)
    const int scol = (tid & 3) * 32;       // element offset within row

    // A-fragment addressing (ISA 16x32 bf16 layout)
    const int mrow = m0 + (lane & 15);
    const int kb   = (lane >> 4) << 3;     // K base 0 or 8

    v8f c0 = {}, c1 = {}, c2 = {}, c3 = {};

    // async-stage one 32xK-slab of B into LDS buffer `buf`
    auto stage = [&](int k0, int buf) {
        const __hip_bfloat16* g = Wt + (size_t)(k0 + srow) * ldw + n0 + scol;
        unsigned      l  = (unsigned)(uintptr_t)&Bsm[buf][srow * BROW + scol];
        unsigned long long ga = (unsigned long long)(uintptr_t)g;
        // inst offset applies to BOTH global and LDS addresses (ISA 10.1/15.18)
        asm volatile("global_load_async_to_lds_b128 %0, %1, off"
                     :: "v"(l), "v"(ga) : "memory");
        asm volatile("global_load_async_to_lds_b128 %0, %1, off offset:16"
                     :: "v"(l), "v"(ga) : "memory");
    };

    stage(0, 0);

    for (int k0 = 0; k0 < K; k0 += 32) {
        const int buf = (k0 >> 5) & 1;

        wait_async0();        // my DMA into Bsm[buf] done
        __syncthreads();      // everyone staged buf; prior compute drained
        if (k0 + 32 < K) stage(k0 + 32, buf ^ 1);   // overlap DMA with WMMAs

        // ---- A fragment: 16x32 bf16 (two 16B global loads) ----
        v16bf a;
        const __hip_bfloat16* pa = X + (size_t)mrow * ldx + (k0 + kb);
        reinterpret_cast<u32x4*>(&a)[0] = *reinterpret_cast<const u32x4*>(pa);       // K=kb..kb+7
        reinterpret_cast<u32x4*>(&a)[1] = *reinterpret_cast<const u32x4*>(pa + 16);  // K=kb+16..kb+23

        // ---- B fragments from LDS: lane holds N-col = lane%16, K = kb..kb+7, kb+16..kb+23 ----
        const __bf16* lb = reinterpret_cast<const __bf16*>(&Bsm[buf][kb * BROW + (lane & 15)]);
        v16bf b;
        for (int i = 0; i < 8; ++i) { b[i] = lb[i * BROW]; b[8 + i] = lb[(16 + i) * BROW]; }
        c0 = __builtin_amdgcn_wmma_f32_16x16x32_bf16(false, a, false, b, (short)0, c0, false, false);

        for (int i = 0; i < 8; ++i) { b[i] = lb[i * BROW + 16]; b[8 + i] = lb[(16 + i) * BROW + 16]; }
        c1 = __builtin_amdgcn_wmma_f32_16x16x32_bf16(false, a, false, b, (short)0, c1, false, false);

        for (int i = 0; i < 8; ++i) { b[i] = lb[i * BROW + 32]; b[8 + i] = lb[(16 + i) * BROW + 32]; }
        c2 = __builtin_amdgcn_wmma_f32_16x16x32_bf16(false, a, false, b, (short)0, c2, false, false);

        for (int i = 0; i < 8; ++i) { b[i] = lb[i * BROW + 48]; b[8 + i] = lb[(16 + i) * BROW + 48]; }
        c3 = __builtin_amdgcn_wmma_f32_16x16x32_bf16(false, a, false, b, (short)0, c3, false, false);
    }

    // ---- D store: VGPR r -> row m0 + r + 8*(lane>=16), col n0 + lane%16 ----
    const int rbase = m0 + ((lane >> 4) << 3);
    const int coff  = n0 + (lane & 15);
#pragma unroll
    for (int r = 0; r < 8; ++r) {
        float* yp = Y + (size_t)(rbase + r) * ldy + coff;
        yp[0]  = c0[r];
        yp[16] = c1[r];
        yp[32] = c2[r];
        yp[48] = c3[r];
    }
}

// ---------------------------------------------------------------------------
// Weight convert + transpose: W[N,K] fp32 -> Wt[K, ldt] bf16 (Wt[k][n]=W[n][k])
// ---------------------------------------------------------------------------
__global__ void convT_kernel(const float* __restrict__ W, __hip_bfloat16* __restrict__ Wt,
                             int N, int K, int ldt) {
    size_t idx = (size_t)blockIdx.x * blockDim.x + threadIdx.x;
    if (idx >= (size_t)N * K) return;
    int k = (int)(idx / N);
    int n = (int)(idx % N);
    Wt[(size_t)k * ldt + n] = __float2bfloat16(W[(size_t)n * K + k]);
}

// ---------------------------------------------------------------------------
// Initial state: input_feed -> x0 feed seg; h0[0]->x0 h seg; h0[1]->x1 h seg;
// c0[0]/c0[1] -> fp32 cell buffers. 64*1024 threads.
// ---------------------------------------------------------------------------
__global__ void init_state_kernel(const float* __restrict__ feed,
                                  const float* __restrict__ h0,
                                  const float* __restrict__ c0,
                                  __hip_bfloat16* __restrict__ x0,
                                  __hip_bfloat16* __restrict__ x1,
                                  float* __restrict__ c0b, float* __restrict__ c1b) {
    int idx = blockIdx.x * blockDim.x + threadIdx.x;     // < 65536
    int b = idx >> 10, j = idx & 1023;
    x0[(size_t)b * 3072 + 1024 + j] = __float2bfloat16(feed[idx]);
    x0[(size_t)b * 3072 + 2048 + j] = __float2bfloat16(h0[idx]);
    x1[(size_t)b * 2048 + 1024 + j] = __float2bfloat16(h0[65536 + idx]);
    c0b[idx] = c0[idx];
    c1b[idx] = c0[65536 + idx];
}

// x0[:, 0:1024] <- bf16(emb[t])
__global__ void pack_emb_kernel(const float* __restrict__ emb_t,
                                __hip_bfloat16* __restrict__ x0) {
    int idx = blockIdx.x * blockDim.x + threadIdx.x;
    int b = idx >> 10, j = idx & 1023;
    x0[(size_t)b * 3072 + j] = __float2bfloat16(emb_t[idx]);
}

// ---------------------------------------------------------------------------
// LSTM pointwise: gates[64,4096] (i,f,g,o blocks of 1024) -> c update, h out.
// h written (bf16) to two destinations for the next GEMM inputs.
// ---------------------------------------------------------------------------
__device__ __forceinline__ float sigmoidf_(float x) { return 1.f / (1.f + __expf(-x)); }

__global__ void lstm_point_kernel(const float* __restrict__ gates,
                                  const float* __restrict__ bi,
                                  const float* __restrict__ bh,
                                  float* __restrict__ c,
                                  __hip_bfloat16* __restrict__ hA, int ldA,
                                  __hip_bfloat16* __restrict__ hB, int ldB) {
    int idx = blockIdx.x * blockDim.x + threadIdx.x;     // < 65536
    int b = idx >> 10, j = idx & 1023;
    const float* g = gates + (size_t)b * 4096;
    float gi = g[j]        + bi[j]        + bh[j];
    float gf = g[1024 + j] + bi[1024 + j] + bh[1024 + j];
    float gg = g[2048 + j] + bi[2048 + j] + bh[2048 + j];
    float go = g[3072 + j] + bi[3072 + j] + bh[3072 + j];
    float i  = sigmoidf_(gi);
    float f  = sigmoidf_(gf);
    float gt = tanhf(gg);
    float o  = sigmoidf_(go);
    float cn = f * c[idx] + i * gt;
    c[idx] = cn;
    float h = o * tanhf(cn);
    __hip_bfloat16 hb = __float2bfloat16(h);
    hA[(size_t)b * ldA + j] = hb;
    hB[(size_t)b * ldB + j] = hb;
}

// ---------------------------------------------------------------------------
// Attention: scores = q . ctx[s,b,:], softmax over S=64, cvec -> xa[:,0:1024],
// align -> output. One block (256 thr) per batch row. ctx layout: [S,B,D].
// ---------------------------------------------------------------------------
__global__ void attn_kernel(const float* __restrict__ q,
                            const float* __restrict__ ctx,
                            float* __restrict__ align_out,
                            __hip_bfloat16* __restrict__ xa) {
    int b   = blockIdx.x;
    int tid = threadIdx.x;                 // 256
    __shared__ float part[4][64];
    __shared__ float sc[64];

    int s   = tid & 63;
    int seg = tid >> 6;
    const float* qb  = q + (size_t)b * DD;
    const float* cs  = ctx + ((size_t)s * BB + b) * DD;
    float acc = 0.f;
#pragma unroll 4
    for (int d = seg * 256; d < seg * 256 + 256; ++d) acc += qb[d] * cs[d];
    part[seg][s] = acc;
    __syncthreads();

    if (tid < 64) sc[tid] = part[0][tid] + part[1][tid] + part[2][tid] + part[3][tid];
    __syncthreads();

    if (tid == 0) {
        float m = -1e30f;
        for (int i = 0; i < 64; ++i) m = fmaxf(m, sc[i]);
        float sum = 0.f;
        for (int i = 0; i < 64; ++i) { float e = __expf(sc[i] - m); sc[i] = e; sum += e; }
        float inv = 1.f / sum;
        for (int i = 0; i < 64; ++i) sc[i] *= inv;
    }
    __syncthreads();

    if (tid < 64) align_out[(size_t)b * SS + tid] = sc[tid];

    for (int d = tid; d < DD; d += 256) {
        float a = 0.f;
#pragma unroll 8
        for (int s2 = 0; s2 < 64; ++s2) a += sc[s2] * ctx[((size_t)s2 * BB + b) * DD + d];
        xa[(size_t)b * 2048 + d] = __float2bfloat16(a);
    }
}

// ---------------------------------------------------------------------------
// attn_h = tanh(ao); write fp32 output[t] and bf16 next-step input-feed.
// ---------------------------------------------------------------------------
__global__ void tanh_feed_kernel(const float* __restrict__ ao,
                                 float* __restrict__ out_t,
                                 __hip_bfloat16* __restrict__ x0) {
    int idx = blockIdx.x * blockDim.x + threadIdx.x;     // < 65536
    int b = idx >> 10, j = idx & 1023;
    float v = tanhf(ao[idx]);
    out_t[idx] = v;
    x0[(size_t)b * 3072 + 1024 + j] = __float2bfloat16(v);
}

// ---------------------------------------------------------------------------
extern "C" void kernel_launch(void* const* d_in, const int* in_sizes, int n_in,
                              void* d_out, int out_size, void* d_ws, size_t ws_size,
                              hipStream_t stream) {
    (void)in_sizes; (void)n_in; (void)out_size; (void)ws_size;

    const float* emb     = (const float*)d_in[0];   // [T,B,E]
    const float* context = (const float*)d_in[1];   // [S,B,D]
    const float* ifeed   = (const float*)d_in[2];   // [1,B,D]
    const float* h0      = (const float*)d_in[3];   // [2,B,D]
    const float* c0      = (const float*)d_in[4];   // [2,B,D]
    const float* W_ih0   = (const float*)d_in[5];   // [4096,2048]
    const float* W_hh0   = (const float*)d_in[6];   // [4096,1024]
    const float* b_ih0   = (const float*)d_in[7];
    const float* b_hh0   = (const float*)d_in[8];
    const float* W_ih1   = (const float*)d_in[9];   // [4096,1024]
    const float* W_hh1   = (const float*)d_in[10];  // [4096,1024]
    const float* b_ih1   = (const float*)d_in[11];
    const float* b_hh1   = (const float*)d_in[12];
    const float* W_in    = (const float*)d_in[13];  // [1024,1024]
    const float* W_out   = (const float*)d_in[14];  // [1024,2048]

    float* out_outputs = (float*)d_out;                               // [T,B,D]
    float* out_attns   = out_outputs + (size_t)TT * BB * DD;          // [T,B,S]

    // ---- carve workspace (256B aligned slices) ----
    char* p = (char*)d_ws;
    auto carve = [&](size_t bytes) -> void* {
        void* r = (void*)p;
        p += (bytes + 255) & ~(size_t)255;
        return r;
    };
    __hip_bfloat16* Wcat0 = (__hip_bfloat16*)carve((size_t)3072 * 4096 * 2); // [K=3072,N=4096]
    __hip_bfloat16* Wcat1 = (__hip_bfloat16*)carve((size_t)2048 * 4096 * 2); // [K=2048,N=4096]
    __hip_bfloat16* Wt_in = (__hip_bfloat16*)carve((size_t)1024 * 1024 * 2); // [1024,1024]
    __hip_bfloat16* Wt_out= (__hip_bfloat16*)carve((size_t)2048 * 1024 * 2); // [2048,1024]
    __hip_bfloat16* x0    = (__hip_bfloat16*)carve((size_t)BB * 3072 * 2);   // [emb|feed|h0]
    __hip_bfloat16* x1    = (__hip_bfloat16*)carve((size_t)BB * 2048 * 2);   // [h0_new|h1_prev]
    __hip_bfloat16* xa    = (__hip_bfloat16*)carve((size_t)BB * 2048 * 2);   // [cvec|h1_new]
    float* gates0 = (float*)carve((size_t)BB * 4096 * 4);
    float* gates1 = (float*)carve((size_t)BB * 4096 * 4);
    float* qbuf   = (float*)carve((size_t)BB * DD * 4);
    float* aobuf  = (float*)carve((size_t)BB * DD * 4);
    float* c0buf  = (float*)carve((size_t)BB * DD * 4);
    float* c1buf  = (float*)carve((size_t)BB * DD * 4);

    const int EB = 256;                       // elementwise block
    const int E64K = (BB * DD) / EB;          // 65536 / 256 = 256 blocks

    // ---- one-time (per call) weight convert + transpose to bf16 [K,N] ----
    {
        size_t n;
        n = (size_t)4096 * 2048;
        convT_kernel<<<(unsigned)((n + EB - 1) / EB), EB, 0, stream>>>(W_ih0, Wcat0, 4096, 2048, 4096);
        n = (size_t)4096 * 1024;
        convT_kernel<<<(unsigned)((n + EB - 1) / EB), EB, 0, stream>>>(W_hh0, Wcat0 + (size_t)2048 * 4096, 4096, 1024, 4096);
        convT_kernel<<<(unsigned)((n + EB - 1) / EB), EB, 0, stream>>>(W_ih1, Wcat1, 4096, 1024, 4096);
        convT_kernel<<<(unsigned)((n + EB - 1) / EB), EB, 0, stream>>>(W_hh1, Wcat1 + (size_t)1024 * 4096, 4096, 1024, 4096);
        n = (size_t)1024 * 1024;
        convT_kernel<<<(unsigned)((n + EB - 1) / EB), EB, 0, stream>>>(W_in, Wt_in, 1024, 1024, 1024);
        n = (size_t)1024 * 2048;
        convT_kernel<<<(unsigned)((n + EB - 1) / EB), EB, 0, stream>>>(W_out, Wt_out, 1024, 2048, 1024);
    }

    // ---- initial state ----
    init_state_kernel<<<E64K, EB, 0, stream>>>(ifeed, h0, c0, x0, x1, c0buf, c1buf);

    // ---- sequential decode over T ----
    for (int t = 0; t < TT; ++t) {
        const float* emb_t = emb + (size_t)t * BB * DD;

        // 1) pack embedding into x0[:, 0:1024]
        pack_emb_kernel<<<E64K, EB, 0, stream>>>(emb_t, x0);

        // 2) layer-0 gates: [64,3072] x [3072,4096]
        wmma_gemm_kernel<<<dim3(4096 / 64), 128, 0, stream>>>(
            x0, 3072, Wcat0, 4096, gates0, 4096, 3072);

        // 3) LSTM0 pointwise: h0 -> x1[:,0:1024] and x0[:,2048:3072]
        lstm_point_kernel<<<E64K, EB, 0, stream>>>(
            gates0, b_ih0, b_hh0, c0buf, x1, 2048, x0 + 2048, 3072);

        // 4) layer-1 gates: [64,2048] x [2048,4096]
        wmma_gemm_kernel<<<dim3(4096 / 64), 128, 0, stream>>>(
            x1, 2048, Wcat1, 4096, gates1, 4096, 2048);

        // 5) LSTM1 pointwise: h1 -> xa[:,1024:2048] and x1[:,1024:2048]
        lstm_point_kernel<<<E64K, EB, 0, stream>>>(
            gates1, b_ih1, b_hh1, c1buf, xa + 1024, 2048, x1 + 1024, 2048);

        // 6) q = h1 @ W_in^T : [64,1024] x [1024,1024] (X = xa second half)
        wmma_gemm_kernel<<<dim3(1024 / 64), 128, 0, stream>>>(
            xa + 1024, 2048, Wt_in, 1024, qbuf, 1024, 1024);

        // 7) attention: scores/softmax/cvec; align -> output; cvec -> xa[:,0:1024]
        attn_kernel<<<BB, 256, 0, stream>>>(
            qbuf, context, out_attns + (size_t)t * BB * SS, xa);

        // 8) attn_h pre-activation: [cvec|h1] @ W_out^T : [64,2048] x [2048,1024]
        wmma_gemm_kernel<<<dim3(1024 / 64), 128, 0, stream>>>(
            xa, 2048, Wt_out, 1024, aobuf, 1024, 2048);

        // 9) tanh -> outputs[t] (fp32) and next input-feed (bf16)
        tanh_feed_kernel<<<E64K, EB, 0, stream>>>(
            aobuf, out_outputs + (size_t)t * BB * DD, x0);
    }
}
